// CasualGraph_77077483094350
// MI455X (gfx1250) — matrix operand order repaired and checked by reading, not compiled, not run on hardware
//
#include <hip/hip_runtime.h>

typedef __attribute__((ext_vector_type(16))) __bf16 v16bf;
typedef __attribute__((ext_vector_type(8)))  __bf16 v8bf;
typedef __attribute__((ext_vector_type(8)))  float  v8f;

#define LN_EPS 1e-5f

__device__ __forceinline__ unsigned short f2bf(float f) {
    unsigned u = __float_as_uint(f);
    u += 0x7FFFu + ((u >> 16) & 1u);          // round-to-nearest-even
    return (unsigned short)(u >> 16);
}

// ---------------------------------------------------------------------------
// T fp32 [N,N] -> TB bf16 row-major  +  TBt bf16 transposed (32x32 LDS tiles)
// ---------------------------------------------------------------------------
__global__ void t_convert_transpose(const float* __restrict__ T,
                                    unsigned short* __restrict__ TB,
                                    unsigned short* __restrict__ TBt, int N) {
    __shared__ unsigned short tile[32][33];
    const int c0 = blockIdx.x * 32;
    const int r0 = blockIdx.y * 32;
    const int x = threadIdx.x, y = threadIdx.y;
    #pragma unroll
    for (int j = 0; j < 4; ++j) {
        const int r = r0 + y + 8 * j;
        const unsigned short b = f2bf(T[(size_t)r * N + c0 + x]);
        TB[(size_t)r * N + c0 + x] = b;
        tile[y + 8 * j][x] = b;
    }
    __syncthreads();
    #pragma unroll
    for (int j = 0; j < 4; ++j) {
        const int c = c0 + y + 8 * j;
        TBt[(size_t)c * N + r0 + x] = tile[x][y + 8 * j];
    }
}

// ---------------------------------------------------------------------------
// H int32 [N,E] -> HBt bf16 [E,N]  (A-matrix of the hyperedge GEMM)
// ---------------------------------------------------------------------------
__global__ void h_convert_transpose(const int* __restrict__ H,
                                    unsigned short* __restrict__ HBt,
                                    int N, int E) {
    __shared__ unsigned short tile[32][33];
    const int e0 = blockIdx.x * 32;
    const int n0 = blockIdx.y * 32;
    const int x = threadIdx.x, y = threadIdx.y;
    #pragma unroll
    for (int j = 0; j < 4; ++j) {
        const int n = n0 + y + 8 * j;
        tile[y + 8 * j][x] =
            (H[(size_t)n * E + e0 + x] > 0) ? (unsigned short)0x3F80 : (unsigned short)0;
    }
    __syncthreads();
    #pragma unroll
    for (int j = 0; j < 4; ++j) {
        const int e = e0 + y + 8 * j;
        HBt[(size_t)e * N + n0 + x] = tile[x][y + 8 * j];
    }
}

__global__ void count_cols(const int* __restrict__ H, float* __restrict__ counts,
                           int N, int E) {
    const int e = blockIdx.x * blockDim.x + threadIdx.x;
    if (e >= E) return;
    float c = 0.f;
    for (int n = 0; n < N; ++n) c += (H[(size_t)n * E + e] > 0) ? 1.f : 0.f;
    counts[e] = c;
}

// ---------------------------------------------------------------------------
// fp32 [R,128] -> bf16 in WMMA B-fragment order: XS[(r/32)*4096 + d*32 + r%32]
// ---------------------------------------------------------------------------
__global__ void convert_swizzle(const float* __restrict__ X,
                                unsigned short* __restrict__ XS, int R) {
    const int t = blockIdx.x * blockDim.x + threadIdx.x;   // R*32 threads
    const int r = t >> 5;
    const int d4 = (t & 31) * 4;
    const float4 v = *(const float4*)(X + (size_t)r * 128 + d4);
    unsigned short* base = XS + (size_t)(r >> 5) * 4096 + (r & 31);
    base[(size_t)(d4 + 0) * 32] = f2bf(v.x);
    base[(size_t)(d4 + 1) * 32] = f2bf(v.y);
    base[(size_t)(d4 + 2) * 32] = f2bf(v.z);
    base[(size_t)(d4 + 3) * 32] = f2bf(v.w);
}

// ---------------------------------------------------------------------------
// Fused LayerNorm (D=128, one wave per row) + bf16 swizzle into XS
// ---------------------------------------------------------------------------
__global__ void layernorm_swizzle(const float* __restrict__ Y,
                                  const float* __restrict__ gamma,
                                  const float* __restrict__ beta,
                                  unsigned short* __restrict__ XS, int R) {
    const int row  = (blockIdx.x * blockDim.x + threadIdx.x) >> 5;
    const int lane = threadIdx.x & 31;
    if (row >= R) return;
    const float4 v = *(const float4*)(Y + (size_t)row * 128 + lane * 4);
    float s = v.x + v.y + v.z + v.w;
    #pragma unroll
    for (int off = 16; off > 0; off >>= 1) s += __shfl_xor(s, off, 32);
    const float m = s * (1.0f / 128.0f);
    const float dx = v.x - m, dy = v.y - m, dz = v.z - m, dw = v.w - m;
    float s2 = dx * dx + dy * dy + dz * dz + dw * dw;
    #pragma unroll
    for (int off = 16; off > 0; off >>= 1) s2 += __shfl_xor(s2, off, 32);
    const float rstd = rsqrtf(s2 * (1.0f / 128.0f) + LN_EPS);
    const float4 g = *(const float4*)(gamma + lane * 4);
    const float4 b = *(const float4*)(beta + lane * 4);
    unsigned short* base = XS + (size_t)(row >> 5) * 4096 + (row & 31);
    const int d4 = lane * 4;
    base[(size_t)(d4 + 0) * 32] = f2bf(dx * rstd * g.x + b.x);
    base[(size_t)(d4 + 1) * 32] = f2bf(dy * rstd * g.y + b.y);
    base[(size_t)(d4 + 2) * 32] = f2bf(dz * rstd * g.z + b.z);
    base[(size_t)(d4 + 3) * 32] = f2bf(dw * rstd * g.w + b.w);
}

// ---------------------------------------------------------------------------
// Out[M,128] f32 = A[M,K] bf16 (row-major, lda) @ XS (pre-swizzled B, 128 wide)
// 256 threads = 8 waves; wave (r,c-half) computes 16 rows x 64 cols,
// grid = M/64 workgroups.  Manual unroll-by-2 ping-pong double buffering:
// compute on one fragment set while the other set's loads are in flight —
// rotation is pure register renaming (no v_mov chains).
// ---------------------------------------------------------------------------
struct Frag {
    v8bf  alo, ahi;
    v16bf b[4];
};

__device__ __forceinline__ void load_frag(Frag& f,
                                          const unsigned short* __restrict__ Arow,
                                          const unsigned short* __restrict__ Bcol,
                                          int k0, int hi) {
    f.alo = *(const v8bf*)(Arow + k0 + hi * 16);
    f.ahi = *(const v8bf*)(Arow + k0 + hi * 16 + 8);
    const unsigned short* nB = Bcol + (size_t)k0 * 128;   // (k0/32)*4096
    #pragma unroll
    for (int c = 0; c < 4; ++c)
        f.b[c] = *(const v16bf*)(nB + (size_t)c * 512);
}

__device__ __forceinline__ void mma_frag(v8f acc[4], const Frag& f) {
    const v16bf a = __builtin_shufflevector(
        f.alo, f.ahi, 0, 1, 2, 3, 4, 5, 6, 7, 8, 9, 10, 11, 12, 13, 14, 15);
    #pragma unroll
    for (int c = 0; c < 4; ++c)
        acc[c] = __builtin_amdgcn_wmma_f32_16x16x32_bf16(
            false, a, false, f.b[c], (short)0, acc[c], false, false);
}

__global__ void __launch_bounds__(256)
gemm_bf16_wmma(const unsigned short* __restrict__ A, int lda,
               const unsigned short* __restrict__ XS,
               float* __restrict__ Out, int K) {
    const int lane  = threadIdx.x & 31;
    const int wave  = threadIdx.x >> 5;
    const int hi    = lane >> 4;          // 0 | 1  (half-wave)
    const int lo    = lane & 15;
    const int chalf = wave & 1;           // column half: 0 -> cols 0..63, 1 -> 64..127

    const int m0 = blockIdx.x * 64 + (wave >> 1) * 16;
    const unsigned short* Arow = A + (size_t)(m0 + lo) * lda;
    const unsigned short* Bcol = XS + (size_t)(chalf * 64 + lo) * 32 + hi * 16;

    v8f acc[4] = {};
    Frag f0, f1;

    const int kt_end = K >> 5;            // even (K = 8192 -> 256)
    load_frag(f0, Arow, Bcol, 0, hi);

    int kt = 0;
    for (; kt + 2 < kt_end; kt += 2) {
        load_frag(f1, Arow, Bcol, (kt + 1) << 5, hi);
        __builtin_prefetch(Arow + ((kt + 1) << 5) + 256, 0, 1);  // global_prefetch_b8
        mma_frag(acc, f0);
        load_frag(f0, Arow, Bcol, (kt + 2) << 5, hi);
        mma_frag(acc, f1);
    }
    // epilogue: kt == kt_end - 2; f0 holds tile kt
    load_frag(f1, Arow, Bcol, (kt + 1) << 5, hi);
    mma_frag(acc, f0);
    mma_frag(acc, f1);

    // C layout: VGPR j -> row (hi*8 + j), col lo of each 16-wide tile
    float* outp = Out + (size_t)m0 * 128 + chalf * 64;
    #pragma unroll
    for (int c = 0; c < 4; ++c) {
        #pragma unroll
        for (int j = 0; j < 8; ++j) {
            outp[(size_t)(hi * 8 + j) * 128 + c * 16 + lo] = acc[c][j];
        }
    }
}

// ---------------------------------------------------------------------------
// out[d] = max_e sums[e,d] / counts[e]
// ---------------------------------------------------------------------------
__global__ void finalize_max(const float* __restrict__ S,
                             const float* __restrict__ counts,
                             float* __restrict__ out, int E) {
    const int d = threadIdx.x;
    float best = -3.402823466e38f;
    for (int e = 0; e < E; ++e)
        best = fmaxf(best, S[(size_t)e * 128 + d] / counts[e]);
    out[d] = best;
}

extern "C" void kernel_launch(void* const* d_in, const int* in_sizes, int n_in,
                              void* d_out, int out_size, void* d_ws, size_t ws_size,
                              hipStream_t stream) {
    (void)n_in; (void)out_size; (void)ws_size;
    const float* x0    = (const float*)d_in[0];
    const float* T     = (const float*)d_in[1];
    const float* gamma = (const float*)d_in[2];
    const float* beta  = (const float*)d_in[3];
    const int*   H     = (const int*)d_in[4];
    float*       outp  = (float*)d_out;

    const int D = 128;
    const int N = in_sizes[0] / D;   // 8192
    const int E = in_sizes[4] / N;   // 4096
    const int LAYERS = 3;            // num_layers from setup_inputs

    char* ws = (char*)d_ws;
    auto alloc = [&](size_t bytes) {
        char* p = ws;
        ws += (bytes + 255) & ~(size_t)255;
        return p;
    };
    unsigned short* TB   = (unsigned short*)alloc((size_t)N * N * 2);
    unsigned short* TBt  = (unsigned short*)alloc((size_t)N * N * 2);
    unsigned short* HBt  = (unsigned short*)alloc((size_t)E * N * 2);
    float*          cnts = (float*)alloc((size_t)E * 4);
    unsigned short* XS   = (unsigned short*)alloc((size_t)N * D * 2);
    unsigned short* TS   = (unsigned short*)alloc((size_t)N * D * 2);
    float*          tmid = (float*)alloc((size_t)N * D * 4);
    float*          ybuf = (float*)alloc((size_t)N * D * 4);
    float*          sums = (float*)alloc((size_t)E * D * 4);

    const dim3 blkT(32, 8);
    t_convert_transpose<<<dim3(N / 32, N / 32), blkT, 0, stream>>>(T, TB, TBt, N);
    h_convert_transpose<<<dim3(E / 32, N / 32), blkT, 0, stream>>>(H, HBt, N, E);
    count_cols<<<(E + 255) / 256, 256, 0, stream>>>(H, cnts, N, E);
    convert_swizzle<<<(N * 32) / 256, 256, 0, stream>>>(x0, XS, N);

    for (int l = 0; l < LAYERS; ++l) {
        gemm_bf16_wmma<<<N / 64, 256, 0, stream>>>(TB, N, XS, tmid, N);
        convert_swizzle<<<(N * 32) / 256, 256, 0, stream>>>(tmid, TS, N);
        gemm_bf16_wmma<<<N / 64, 256, 0, stream>>>(TBt, N, TS, ybuf, N);
        layernorm_swizzle<<<(N * 32) / 256, 256, 0, stream>>>(ybuf, gamma, beta, XS, N);
    }

    gemm_bf16_wmma<<<E / 64, 256, 0, stream>>>(HBt, N, XS, sums, N);
    finalize_max<<<1, 128, 0, stream>>>(sums, cnts, outp, E);
}
